// TransformerBlock_50637664420069
// MI455X (gfx1250) — compile-verified, matrix operand order checked by reading, not run on hardware
//
#include <hip/hip_runtime.h>

// ---------------------------------------------------------------- constants
static constexpr int kD  = 1024;
static constexpr int kH  = 16;
static constexpr int kHD = 64;
static constexpr int kNL = 6;
static constexpr int kDF = 4096;
static constexpr int kV  = 32000;
static constexpr int kB  = 2;
static constexpr int kL  = 1024;
static constexpr int kBL = kB * kL;   // 2048 rows

typedef __attribute__((ext_vector_type(16))) _Float16 v16h;
typedef __attribute__((ext_vector_type(8)))  float    v8f;

union AFrag { v16h h; unsigned u[8]; };

__device__ __forceinline__ unsigned pack2(_Float16 a, _Float16 b) {
  union { _Float16 h[2]; unsigned u; } t; t.h[0] = a; t.h[1] = b; return t.u;
}

// K index of the first element held in 32-bit fragment register v for this
// lane-half, per ISA 7.12.2 (16-bit A 16x32 layout).
__device__ __forceinline__ int kform(int v, int half8) {
  return ((v & 4) << 2) | half8 | ((v & 3) << 1);
}

__device__ __forceinline__ v8f wmma_f16(v16h a, v16h b, v8f c) {
  return __builtin_amdgcn_wmma_f32_16x16x32_f16(false, a, false, b,
                                                (short)0, c, false, false);
}

// 16-byte async global->LDS copy, tracked by ASYNCcnt (CDNA5 §10 async ops).
// Low 32 bits of a generic LDS pointer are the LDS byte address (ISA 10.2
// aperture rule).
__device__ __forceinline__ void async_cp16(const _Float16* g, _Float16* l) {
  unsigned lds = (unsigned)(unsigned long long)l;
  unsigned long long ga = (unsigned long long)g;
  asm volatile("global_load_async_to_lds_b128 %0, %1, off"
               :: "v"(lds), "v"(ga) : "memory");
}
__device__ __forceinline__ void wait_async0() {
  asm volatile("s_wait_asynccnt 0x0" ::: "memory");
}

// ---------------------------------------------------------------- GEMM
// C[M,N] = A16[M,K] @ W16[N,K]^T (+bias[n]) (+ReLU), f16 in / f32 out.
// M%128==0, N%128==0, K%32==0. Double-buffered async LDS staging.
#define BT_M 128
#define BT_N 128
#define BT_K 32
#define LDH  40   // f16 per LDS row: 80 B, keeps 16 B alignment for b128 fills

__global__ __launch_bounds__(256)
void gemm_wmma_kernel(const _Float16* __restrict__ A, const _Float16* __restrict__ W,
                      const float* __restrict__ bias, float* __restrict__ C,
                      int M, int N, int K, int relu) {
  __shared__ _Float16 As[2][BT_M * LDH];
  __shared__ _Float16 Bs[2][BT_N * LDH];

  const int t      = threadIdx.x;
  const int lane   = t & 31;
  const int wave   = t >> 5;        // 0..7
  const int waveM  = wave >> 1;     // 0..3 -> 32-row slab
  const int waveN  = wave & 1;      // 0..1 -> 64-col slab
  const int blockM = blockIdx.y * BT_M;
  const int blockN = blockIdx.x * BT_N;
  const int l15    = lane & 15;
  const int half8  = (lane >> 4) << 3;

  // async-fill mapping: 16 B (8 f16) chunks, 4 chunks per 32-f16 row.
  const int r0  = t >> 2;           // 0..63 (rows r0 and r0+64)
  const int seg = (t & 3) * 8;      // f16 offset within row

  v8f acc[2][4] = {};

  auto issue = [&](int buf, int k0) {
    async_cp16(&A[(size_t)(blockM + r0)      * K + k0 + seg], &As[buf][ r0       * LDH + seg]);
    async_cp16(&A[(size_t)(blockM + r0 + 64) * K + k0 + seg], &As[buf][(r0 + 64) * LDH + seg]);
    async_cp16(&W[(size_t)(blockN + r0)      * K + k0 + seg], &Bs[buf][ r0       * LDH + seg]);
    async_cp16(&W[(size_t)(blockN + r0 + 64) * K + k0 + seg], &Bs[buf][(r0 + 64) * LDH + seg]);
  };

  issue(0, 0);

  for (int k0 = 0; k0 < K; k0 += BT_K) {
    const int cur = (k0 >> 5) & 1;
    wait_async0();                   // this tile's fills complete
    __syncthreads();
    if (k0 + BT_K < K) {
      issue(cur ^ 1, k0 + BT_K);     // overlap next tile's fills with compute
      if (k0 + 2 * BT_K < K)         // and hint the tile after that into L2
        __builtin_prefetch(&A[(size_t)(blockM + r0) * K + k0 + 2 * BT_K], 0, 3);
    }

    AFrag af[2], bf[4];
#pragma unroll
    for (int mt = 0; mt < 2; ++mt) {
      const int row = waveM * 32 + mt * 16 + l15;
#pragma unroll
      for (int v = 0; v < 8; ++v)
        af[mt].u[v] = *(const unsigned*)&As[cur][row * LDH + kform(v, half8)];
    }
#pragma unroll
    for (int nt = 0; nt < 4; ++nt) {
      const int row = waveN * 64 + nt * 16 + l15;
#pragma unroll
      for (int v = 0; v < 8; ++v)
        bf[nt].u[v] = *(const unsigned*)&Bs[cur][row * LDH + kform(v, half8)];
    }
#pragma unroll
    for (int mt = 0; mt < 2; ++mt)
#pragma unroll
      for (int nt = 0; nt < 4; ++nt)
        acc[mt][nt] = wmma_f16(af[mt].h, bf[nt].h, acc[mt][nt]);
    __syncthreads();
  }

  // epilogue: C layout = rows in VGPRs (r, r+8 by lane-half), cols in lanes
#pragma unroll
  for (int mt = 0; mt < 2; ++mt)
#pragma unroll
    for (int nt = 0; nt < 4; ++nt) {
      const int n = blockN + waveN * 64 + nt * 16 + l15;
      const float bv = bias ? bias[n] : 0.0f;
#pragma unroll
      for (int r = 0; r < 8; ++r) {
        const int m = blockM + waveM * 32 + mt * 16 + r + half8;
        float val = acc[mt][nt][r] + bv;
        if (relu) val = fmaxf(val, 0.0f);
        C[(size_t)m * N + n] = val;
      }
    }
}

// ---------------------------------------------------------------- fp32->f16
__global__ __launch_bounds__(256)
void cvt_f16_kernel(const float* __restrict__ in, _Float16* __restrict__ out) {
  int i = blockIdx.x * 1024 + threadIdx.x;   // total size is a multiple of 1024
#pragma unroll
  for (int j = 0; j < 4; ++j, i += 256) out[i] = (_Float16)in[i];
}

// ---------------------------------------------------------------- attention
// One wave per (b, h, 16-query tile). Flash-style online softmax with WMMA
// for both Q*K^T and P*V; width-16 shuffles match the C-fragment row split.
#define PS_STRIDE 36

__global__ __launch_bounds__(32)
void attn_wmma_kernel(const float* __restrict__ Q, const float* __restrict__ Km,
                      const float* __restrict__ Vm, float* __restrict__ O,
                      const int* __restrict__ padtok, int causal) {
  const int lane  = threadIdx.x;
  const int l15   = lane & 15;
  const int half8 = (lane >> 4) << 3;
  const int m0    = blockIdx.x * 16;
  const int h     = blockIdx.y;
  const int b     = blockIdx.z;

  const float* Qp = Q  + (size_t)b * kL * kD + h * kHD;
  const float* Kp = Km + (size_t)b * kL * kD + h * kHD;
  const float* Vp = Vm + (size_t)b * kL * kD + h * kHD;
  const int*  tok = padtok + (size_t)b * kL;

  __shared__ _Float16 Ps[16 * PS_STRIDE];

  AFrag qf[2];
  {
    const float* qp = Qp + (size_t)(m0 + l15) * kD;
#pragma unroll
    for (int v = 0; v < 8; ++v) {
      const int d = kform(v, half8);
      qf[0].u[v] = pack2((_Float16)qp[d],      (_Float16)qp[d + 1]);
      qf[1].u[v] = pack2((_Float16)qp[32 + d], (_Float16)qp[32 + d + 1]);
    }
  }

  float mi[8], li[8];
#pragma unroll
  for (int r = 0; r < 8; ++r) { mi[r] = -1e30f; li[r] = 0.0f; }
  v8f o[4] = {};

  for (int j0 = 0; j0 < kL; j0 += 32) {
    v8f s[2];
#pragma unroll
    for (int nt = 0; nt < 2; ++nt) {
      v8f sa = {};
      const float* kp = Kp + (size_t)(j0 + nt * 16 + l15) * kD;
      AFrag kf0, kf1;
#pragma unroll
      for (int v = 0; v < 8; ++v) {
        const int d = kform(v, half8);
        kf0.u[v] = pack2((_Float16)kp[d],      (_Float16)kp[d + 1]);
        kf1.u[v] = pack2((_Float16)kp[32 + d], (_Float16)kp[32 + d + 1]);
      }
      sa = wmma_f16(qf[0].h, kf0.h, sa);
      sa = wmma_f16(qf[1].h, kf1.h, sa);
      s[nt] = sa;
    }

#pragma unroll
    for (int nt = 0; nt < 2; ++nt) {
      const int key = j0 + nt * 16 + l15;
      const bool pad = (tok[key] == 0);
#pragma unroll
      for (int r = 0; r < 8; ++r) {
        float sv = s[nt][r] * 0.125f;          // 1/sqrt(64)
        const int qr = m0 + r + half8;
        if (pad || (causal && key > qr)) sv = -1e30f;
        s[nt][r] = sv;
      }
    }

    float alpha[8];
#pragma unroll
    for (int r = 0; r < 8; ++r) {
      float v = fmaxf(s[0][r], s[1][r]);
#pragma unroll
      for (int off = 8; off > 0; off >>= 1)
        v = fmaxf(v, __shfl_xor(v, off, 16));
      const float mnew = fmaxf(mi[r], v);
      alpha[r] = __expf(mi[r] - mnew);
      mi[r] = mnew;
      const float p0 = __expf(s[0][r] - mnew);
      const float p1 = __expf(s[1][r] - mnew);
      s[0][r] = p0; s[1][r] = p1;
      float ps = p0 + p1;
#pragma unroll
      for (int off = 8; off > 0; off >>= 1)
        ps += __shfl_xor(ps, off, 16);
      li[r] = li[r] * alpha[r] + ps;
    }
#pragma unroll
    for (int tt = 0; tt < 4; ++tt)
#pragma unroll
      for (int r = 0; r < 8; ++r) o[tt][r] *= alpha[r];

    // repack P (C layout) -> LDS -> A fragment (single wave: DS is in-order)
#pragma unroll
    for (int nt = 0; nt < 2; ++nt)
#pragma unroll
      for (int r = 0; r < 8; ++r)
        Ps[(r + half8) * PS_STRIDE + nt * 16 + l15] = (_Float16)s[nt][r];
    __syncthreads();
    AFrag pf;
#pragma unroll
    for (int v = 0; v < 8; ++v)
      pf.u[v] = *(const unsigned*)&Ps[l15 * PS_STRIDE + kform(v, half8)];

#pragma unroll
    for (int tt = 0; tt < 4; ++tt) {
      AFrag vf;
      const int d = tt * 16 + l15;
#pragma unroll
      for (int v = 0; v < 8; ++v) {
        const int key = j0 + kform(v, half8);
        vf.u[v] = pack2((_Float16)Vp[(size_t)key * kD + d],
                        (_Float16)Vp[(size_t)(key + 1) * kD + d]);
      }
      o[tt] = wmma_f16(pf.h, vf.h, o[tt]);
    }
    __syncthreads();
  }

#pragma unroll
  for (int r = 0; r < 8; ++r) {
    const int row = m0 + r + half8;
    const float inv = 1.0f / fmaxf(li[r], 1e-20f);
#pragma unroll
    for (int tt = 0; tt < 4; ++tt)
      O[((size_t)b * kL + row) * kD + h * kHD + tt * 16 + l15] = o[tt][r] * inv;
  }
}

// ---------------------------------------------------------------- LN(x+res)
__global__ __launch_bounds__(256)
void ln_residual_kernel(const float* __restrict__ X, const float* __restrict__ R,
                        const float* __restrict__ g, const float* __restrict__ be,
                        float* __restrict__ Out) {
  const int row = blockIdx.x;
  const int tid = threadIdx.x;
  const float* xp = X + (size_t)row * kD;
  const float* rp = R + (size_t)row * kD;
  __shared__ float red[256];

  float vals[4];
  float s = 0.0f;
#pragma unroll
  for (int i = 0; i < 4; ++i) {
    const int c = tid + i * 256;
    vals[i] = xp[c] + rp[c];
    s += vals[i];
  }
  red[tid] = s; __syncthreads();
  for (int off = 128; off > 0; off >>= 1) {
    if (tid < off) red[tid] += red[tid + off];
    __syncthreads();
  }
  const float mean = red[0] * (1.0f / kD);
  __syncthreads();

  float vs = 0.0f;
#pragma unroll
  for (int i = 0; i < 4; ++i) { const float d = vals[i] - mean; vs += d * d; }
  red[tid] = vs; __syncthreads();
  for (int off = 128; off > 0; off >>= 1) {
    if (tid < off) red[tid] += red[tid + off];
    __syncthreads();
  }
  const float inv = rsqrtf(red[0] * (1.0f / kD) + 1e-5f);
#pragma unroll
  for (int i = 0; i < 4; ++i) {
    const int c = tid + i * 256;
    Out[(size_t)row * kD + c] = (vals[i] - mean) * inv * g[c] + be[c];
  }
}

// ---------------------------------------------------------------- embedding
__global__ __launch_bounds__(256)
void embed_kernel(const int* __restrict__ tok, const float* __restrict__ emb,
                  float* __restrict__ out) {
  const int row = blockIdx.x;
  const int id  = tok[row];
  for (int c = threadIdx.x; c < kD; c += 256)
    out[(size_t)row * kD + c] = emb[(size_t)id * kD + c] * 32.0f; // sqrt(1024)
}

// ---------------------------------------------------------------- softmax(V)
__global__ __launch_bounds__(256)
void softmax_rows_kernel(float* __restrict__ Y) {
  const int row = blockIdx.x;
  const int tid = threadIdx.x;
  float* p = Y + (size_t)row * kV;
  __shared__ float red[256];

  float mx = -1e30f;
  for (int c = tid; c < kV; c += 256) mx = fmaxf(mx, p[c]);
  red[tid] = mx; __syncthreads();
  for (int off = 128; off > 0; off >>= 1) {
    if (tid < off) red[tid] = fmaxf(red[tid], red[tid + off]);
    __syncthreads();
  }
  mx = red[0]; __syncthreads();

  float s = 0.0f;
  for (int c = tid; c < kV; c += 256) {
    const float e = __expf(p[c] - mx);
    p[c] = e; s += e;
  }
  red[tid] = s; __syncthreads();
  for (int off = 128; off > 0; off >>= 1) {
    if (tid < off) red[tid] += red[tid + off];
    __syncthreads();
  }
  const float inv = 1.0f / red[0];
  for (int c = tid; c < kV; c += 256) p[c] *= inv;
}

// ---------------------------------------------------------------- host side
struct Scratch { _Float16 *A16, *W16; };

static inline void gemm(const float* A, const float* W, const float* bias,
                        float* C, int M, int N, int K, int relu,
                        const Scratch& sc, hipStream_t s) {
  cvt_f16_kernel<<<(unsigned)((size_t)M * K / 1024), 256, 0, s>>>(A, sc.A16);
  cvt_f16_kernel<<<(unsigned)((size_t)N * K / 1024), 256, 0, s>>>(W, sc.W16);
  dim3 grid((unsigned)(N / BT_N), (unsigned)(M / BT_M));
  gemm_wmma_kernel<<<grid, 256, 0, s>>>(sc.A16, sc.W16, bias, C, M, N, K, relu);
}

extern "C" void kernel_launch(void* const* d_in, const int* in_sizes, int n_in,
                              void* d_out, int out_size, void* d_ws, size_t ws_size,
                              hipStream_t stream) {
  (void)in_sizes; (void)n_in; (void)out_size; (void)ws_size;

  const int*   src   = (const int*)  d_in[0];
  const int*   tgt   = (const int*)  d_in[1];
  const float* emb   = (const float*)d_in[2];
  const float* out_w = (const float*)d_in[3];
  const float* ewq   = (const float*)d_in[4];
  const float* ewk   = (const float*)d_in[5];
  const float* ewv   = (const float*)d_in[6];
  const float* ewo   = (const float*)d_in[7];
  const float* ew1   = (const float*)d_in[8];
  const float* eb1   = (const float*)d_in[9];
  const float* ew2   = (const float*)d_in[10];
  const float* eb2   = (const float*)d_in[11];
  const float* eg1   = (const float*)d_in[12];
  const float* ebe1  = (const float*)d_in[13];
  const float* eg2   = (const float*)d_in[14];
  const float* ebe2  = (const float*)d_in[15];
  const float* swq   = (const float*)d_in[16];
  const float* swk   = (const float*)d_in[17];
  const float* swv   = (const float*)d_in[18];
  const float* swo   = (const float*)d_in[19];
  const float* cwq   = (const float*)d_in[20];
  const float* cwk   = (const float*)d_in[21];
  const float* cwv   = (const float*)d_in[22];
  const float* cwo   = (const float*)d_in[23];
  const float* dw1   = (const float*)d_in[24];
  const float* db1   = (const float*)d_in[25];
  const float* dw2   = (const float*)d_in[26];
  const float* db2   = (const float*)d_in[27];
  const float* dg1   = (const float*)d_in[28];
  const float* dbe1  = (const float*)d_in[29];
  const float* dg2   = (const float*)d_in[30];
  const float* dbe2  = (const float*)d_in[31];
  const float* dg3   = (const float*)d_in[32];
  const float* dbe3  = (const float*)d_in[33];

  // workspace layout (all offsets multiples of 16 B)
  const size_t MD = (size_t)kBL * kD;         // 2M floats
  float* ws = (float*)d_ws;
  float* X  = ws;
  float* T1 = X  + MD;
  float* T2 = T1 + MD;
  float* Qb = T2 + MD;
  float* Kb = Qb + MD;
  float* Vb = Kb + MD;
  float* AO = Vb + MD;
  float* F  = AO + MD;                        // [kBL, kDF]
  float* endf = F + (size_t)kBL * kDF;
  Scratch sc;
  sc.A16 = (_Float16*)endf;                                   // max 2048*4096 f16
  sc.W16 = sc.A16 + (size_t)kBL * kDF;                        // max 32000*1024 f16

  const dim3 attnGrid((unsigned)(kL / 16), (unsigned)kH, (unsigned)kB);
  const size_t DD = (size_t)kD * kD;

  embed_kernel<<<kBL, 256, 0, stream>>>(src, emb, X);

  // -------- encoder: 6 layers
  for (int i = 0; i < kNL; ++i) {
    gemm(X, ewq + i * DD, nullptr, Qb, kBL, kD, kD, 0, sc, stream);
    gemm(X, ewk + i * DD, nullptr, Kb, kBL, kD, kD, 0, sc, stream);
    gemm(X, ewv + i * DD, nullptr, Vb, kBL, kD, kD, 0, sc, stream);
    attn_wmma_kernel<<<attnGrid, 32, 0, stream>>>(Qb, Kb, Vb, AO, src, 0);
    gemm(AO, ewo + i * DD, nullptr, T1, kBL, kD, kD, 0, sc, stream);
    ln_residual_kernel<<<kBL, 256, 0, stream>>>(X, T1, eg1 + i * kD, ebe1 + i * kD, X);
    gemm(X, ew1 + (size_t)i * kDF * kD, eb1 + (size_t)i * kDF, F, kBL, kDF, kD, 1, sc, stream);
    gemm(F, ew2 + (size_t)i * kD * kDF, eb2 + (size_t)i * kD, T1, kBL, kD, kDF, 0, sc, stream);
    ln_residual_kernel<<<kBL, 256, 0, stream>>>(X, T1, eg2 + i * kD, ebe2 + i * kD, X);
  }
  // X now holds encoder memory.

  // -------- decoder: the reference overwrites y each iteration and feeds
  // `memory` into every layer, so the output depends only on layer NL-1.
  {
    const int i = kNL - 1;
    gemm(X, swq + i * DD, nullptr, Qb, kBL, kD, kD, 0, sc, stream);
    gemm(X, swk + i * DD, nullptr, Kb, kBL, kD, kD, 0, sc, stream);
    gemm(X, swv + i * DD, nullptr, Vb, kBL, kD, kD, 0, sc, stream);
    attn_wmma_kernel<<<attnGrid, 32, 0, stream>>>(Qb, Kb, Vb, AO, tgt, 1);
    gemm(AO, swo + i * DD, nullptr, T1, kBL, kD, kD, 0, sc, stream);
    ln_residual_kernel<<<kBL, 256, 0, stream>>>(X, T1, dg1 + i * kD, dbe1 + i * kD, T2);
    // cross-attn (reference bug): Q = mem@cwq^T, K = mem@cwk^T, V = t@cwv^T
    gemm(X,  cwq + i * DD, nullptr, Qb, kBL, kD, kD, 0, sc, stream);
    gemm(X,  cwk + i * DD, nullptr, Kb, kBL, kD, kD, 0, sc, stream);
    gemm(T2, cwv + i * DD, nullptr, Vb, kBL, kD, kD, 0, sc, stream);
    attn_wmma_kernel<<<attnGrid, 32, 0, stream>>>(Qb, Kb, Vb, AO, src, 0);
    gemm(AO, cwo + i * DD, nullptr, T1, kBL, kD, kD, 0, sc, stream);
    ln_residual_kernel<<<kBL, 256, 0, stream>>>(T2, T1, dg2 + i * kD, dbe2 + i * kD, T2);
    gemm(T2, dw1 + (size_t)i * kDF * kD, db1 + (size_t)i * kDF, F, kBL, kDF, kD, 1, sc, stream);
    gemm(F,  dw2 + (size_t)i * kD * kDF, db2 + (size_t)i * kD, T1, kBL, kD, kDF, 0, sc, stream);
    ln_residual_kernel<<<kBL, 256, 0, stream>>>(T2, T1, dg3 + i * kD, dbe3 + i * kD, T2);
  }

  // -------- logits + softmax over V
  gemm(T2, out_w, nullptr, (float*)d_out, kBL, kV, kD, 0, sc, stream);
  softmax_rows_kernel<<<kBL, 256, 0, stream>>>((float*)d_out);
}
